// GCN_3l_Gelu_37787122270456
// MI455X (gfx1250) — matrix operand order repaired and checked by reading, not compiled, not run on hardware
//
#include <hip/hip_runtime.h>
#include <math.h>

// ---------------------------------------------------------------------------
// Types for WMMA (gfx1250, wave32)
// ---------------------------------------------------------------------------
typedef __attribute__((ext_vector_type(16))) __bf16       v16bf;
typedef __attribute__((ext_vector_type(8)))  float        v8f;
typedef __attribute__((ext_vector_type(4)))  unsigned int u32x4;
typedef __attribute__((ext_vector_type(4)))  float        f32x4;

union ABPack { u32x4 u[2]; v16bf v; };

__device__ __forceinline__ __bf16 f2bf(float x) {
  // round-to-nearest-even fp32 -> bf16
  unsigned u = __float_as_uint(x);
  unsigned r = (u + 0x7FFFu + ((u >> 16) & 1u)) >> 16;
  unsigned short hs = (unsigned short)r;
  __bf16 b;
  __builtin_memcpy(&b, &hs, 2);
  return b;
}

// ---------------------------------------------------------------------------
// Conversions
// ---------------------------------------------------------------------------
__global__ void k_f32_to_bf16(const float* __restrict__ in, __bf16* __restrict__ out, int n) {
  int i = blockIdx.x * blockDim.x + threadIdx.x;
  if (i < n) out[i] = f2bf(in[i]);
}

// Wf is [10][64]; pad to [16][64] with zeros so the WMMA N-tile is full.
__global__ void k_wf_pad_bf16(const float* __restrict__ in, __bf16* __restrict__ out) {
  int i = blockIdx.x * blockDim.x + threadIdx.x;  // 1024 threads total
  if (i >= 16 * 64) return;
  int r = i >> 6, c = i & 63;
  out[i] = (r < 10) ? f2bf(in[r * 64 + c]) : f2bf(0.0f);
}

// ---------------------------------------------------------------------------
// Degree / normalization:  deg[i] = in-degree(i) + 1 (self loop); dinv = rsqrt
// ---------------------------------------------------------------------------
__global__ void k_deg_init(float* __restrict__ deg, int N) {
  int i = blockIdx.x * blockDim.x + threadIdx.x;
  if (i < N) deg[i] = 1.0f;  // self-loop
}

__global__ void k_deg_count(const int* __restrict__ dst, float* __restrict__ deg, int E) {
  int i = blockIdx.x * blockDim.x + threadIdx.x;
  if (i < E) atomicAdd(&deg[dst[i]], 1.0f);
}

__global__ void k_deg_rsqrt(float* __restrict__ deg, int N) {
  int i = blockIdx.x * blockDim.x + threadIdx.x;
  if (i < N) deg[i] = rsqrtf(deg[i]);
}

// ---------------------------------------------------------------------------
// GEMM: C[N x NCOLS] = A[N x DIN](bf16) * W[NTILES*16 x DIN](bf16)^T  (+bias)
//
// One wave owns a 16-row x (NTILES*16)-col output strip: the A fragment is
// loaded ONCE per K-step and reused across NTILES WMMAs (4x less A traffic
// than one-tile-per-wave; B is tiny and L0/L2 resident for every block).
// block = 256 threads = 8 waves = 8 row tiles; grid.x = ceil(rowTiles/8).
// ---------------------------------------------------------------------------
template <int DIN, int NTILES, int LDC, int NCOLS, bool BIAS>
__global__ void k_gemm_bf16(const __bf16* __restrict__ A,
                            const __bf16* __restrict__ W,
                            const float* __restrict__ bias,
                            float* __restrict__ C, int N) {
  const int lane    = threadIdx.x & 31;
  const int wave    = threadIdx.x >> 5;
  const int rowTile = blockIdx.x * 8 + wave;
  const int half    = lane >> 4;               // 0/1
  const int l15     = lane & 15;
  int m = rowTile * 16 + l15;                  // A row this lane loads
  if (m >= N) m = N - 1;                       // clamp; stores are guarded
  const int kA = half * 8;                     // A: k-base within 32-chunk
  const int kB = half * 8;                     // B: k-base within 32-chunk

  v8f acc[NTILES];
#pragma unroll
  for (int t = 0; t < NTILES; ++t) acc[t] = (v8f){};

#pragma unroll
  for (int k0 = 0; k0 < DIN; k0 += 32) {
    ABPack a;
    a.u[0] = *(const u32x4*)(A + (size_t)m * DIN + k0 + kA);
    a.u[1] = *(const u32x4*)(A + (size_t)m * DIN + k0 + kA + 16);
#pragma unroll
    for (int t = 0; t < NTILES; ++t) {
      const int n = t * 16 + l15;              // W row == output column
      ABPack b;
      b.u[0] = *(const u32x4*)(W + (size_t)n * DIN + k0 + kB);
      b.u[1] = *(const u32x4*)(W + (size_t)n * DIN + k0 + kB + 16);
      acc[t] = __builtin_amdgcn_wmma_f32_16x16x32_bf16(false, a.v, false, b.v,
                                                       (short)0, acc[t], false, false);
    }
  }

  // D layout: VGPR r -> M = half*8 + r, N = l15 (per ISA 32-bit C/D table)
  const int mbase = rowTile * 16 + half * 8;
#pragma unroll
  for (int t = 0; t < NTILES; ++t) {
    const int col = t * 16 + l15;
    if (col >= NCOLS) continue;
    const float bv = BIAS ? bias[col] : 0.0f;
#pragma unroll
    for (int r = 0; r < 8; ++r) {
      const int mm = mbase + r;
      if (mm < N) C[(size_t)mm * LDC + col] = acc[t][r] + bv;
    }
  }
}

// ---------------------------------------------------------------------------
// Aggregation:  agg = dinv[i]^2 * h[i] + bias      (self-loop + bias init)
//               agg[dst] += dinv[src]*dinv[dst]*h[src]   (edges, f32 atomics)
// ---------------------------------------------------------------------------
__global__ void k_agg_init(const float* __restrict__ h, const float* __restrict__ dinv,
                           const float* __restrict__ bias, float* __restrict__ agg, int N) {
  int gid = blockIdx.x * blockDim.x + threadIdx.x;  // N*16 threads, 4 dims each
  if (gid >= N * 16) return;
  int i  = gid >> 4;
  int d4 = (gid & 15) * 4;
  float di = dinv[i];
  float s  = di * di;
  f32x4 hv = *(const f32x4*)(h + (size_t)i * 64 + d4);
  f32x4 bv = *(const f32x4*)(bias + d4);
  f32x4 o  = hv * s + bv;
  *(f32x4*)(agg + (size_t)i * 64 + d4) = o;
}

__global__ void k_edge_agg(const int* __restrict__ src, const int* __restrict__ dst,
                           const float* __restrict__ dinv, const float* __restrict__ h,
                           float* __restrict__ agg, int E) {
  long long gid = (long long)blockIdx.x * blockDim.x + threadIdx.x;  // E*16 threads
  int e = (int)(gid >> 4);
  if (e >= E) return;
  int d4 = ((int)gid & 15) * 4;
  int s = src[e], t = dst[e];
  float nrm = dinv[s] * dinv[t];
  f32x4 hv = *(const f32x4*)(h + (size_t)s * 64 + d4);
  float* p = agg + (size_t)t * 64 + d4;
  atomicAdd(p + 0, hv.x * nrm);
  atomicAdd(p + 1, hv.y * nrm);
  atomicAdd(p + 2, hv.z * nrm);
  atomicAdd(p + 3, hv.w * nrm);
}

// ---------------------------------------------------------------------------
// BatchNorm (training stats) + exact GELU, output bf16 for next WMMA layer
// ---------------------------------------------------------------------------
__global__ void k_zero128(float* __restrict__ s) { s[threadIdx.x] = 0.0f; }

__global__ void k_bn_stats(const float* __restrict__ agg, float* __restrict__ stats, int N) {
  __shared__ float ss[256], ss2[256];
  int t   = threadIdx.x;
  int f   = t & 63;
  int rep = t >> 6;                 // 4 replicas per feature
  int i0  = blockIdx.x * 128;       // 128 rows per block
  float s = 0.0f, s2 = 0.0f;
  for (int r = rep; r < 128; r += 4) {
    int i = i0 + r;
    if (i < N) {
      float v = agg[(size_t)i * 64 + f];
      s  += v;
      s2 += v * v;
    }
  }
  ss[t] = s; ss2[t] = s2;
  __syncthreads();
  if (t < 64) {
    s  = ss[t]  + ss[t + 64]  + ss[t + 128]  + ss[t + 192];
    s2 = ss2[t] + ss2[t + 64] + ss2[t + 128] + ss2[t + 192];
    atomicAdd(&stats[t], s);
    atomicAdd(&stats[64 + t], s2);
  }
}

__global__ void k_bn_gelu(const float* __restrict__ agg, const float* __restrict__ stats,
                          const float* __restrict__ gamma, const float* __restrict__ beta,
                          __bf16* __restrict__ hb, int N) {
  int gid = blockIdx.x * blockDim.x + threadIdx.x;  // N*64
  if (gid >= N * 64) return;
  int f = gid & 63;
  float invN  = 1.0f / (float)N;
  float mean  = stats[f] * invN;
  float var   = stats[64 + f] * invN - mean * mean;
  float scale = rsqrtf(var + 1e-5f) * gamma[f];
  float shift = beta[f] - mean * scale;
  float hn = agg[gid] * scale + shift;
  float g  = 0.5f * hn * (1.0f + erff(hn * 0.70710678118654752f));
  hb[gid] = f2bf(g);
}

// ---------------------------------------------------------------------------
// Host orchestration
// ---------------------------------------------------------------------------
extern "C" void kernel_launch(void* const* d_in, const int* in_sizes, int n_in,
                              void* d_out, int out_size, void* d_ws, size_t ws_size,
                              hipStream_t stream) {
  (void)n_in; (void)out_size; (void)ws_size;

  const float* x   = (const float*)d_in[0];
  const int*  edge = (const int*)d_in[1];
  const float* W1  = (const float*)d_in[2];
  const float* b1  = (const float*)d_in[3];
  const float* g1  = (const float*)d_in[4];
  const float* be1 = (const float*)d_in[5];
  const float* W2  = (const float*)d_in[6];
  const float* b2  = (const float*)d_in[7];
  const float* g2  = (const float*)d_in[8];
  const float* be2 = (const float*)d_in[9];
  const float* W3  = (const float*)d_in[10];
  const float* b3  = (const float*)d_in[11];
  const float* g3  = (const float*)d_in[12];
  const float* be3 = (const float*)d_in[13];
  const float* Wf  = (const float*)d_in[14];
  const float* bf  = (const float*)d_in[15];
  float* out = (float*)d_out;

  const int N = in_sizes[0] / 128;   // 100000
  const int E = in_sizes[1] / 2;     // 1600000
  const int* src = edge;
  const int* dst = edge + E;

  // --- carve workspace ---
  char* ws = (char*)d_ws;
  size_t cur = 0;
  auto carve = [&](size_t bytes) -> void* {
    cur = (cur + 255) & ~(size_t)255;
    void* p = (void*)(ws + cur);
    cur += bytes;
    return p;
  };
  float*  dinv  = (float*)carve((size_t)N * 4);
  float*  h     = (float*)carve((size_t)N * 64 * 4);
  float*  agg   = (float*)carve((size_t)N * 64 * 4);
  __bf16* hb    = (__bf16*)carve((size_t)N * 128 * 2);
  __bf16* w1b   = (__bf16*)carve((size_t)64 * 128 * 2);
  __bf16* w2b   = (__bf16*)carve((size_t)64 * 64 * 2);
  __bf16* w3b   = (__bf16*)carve((size_t)64 * 64 * 2);
  __bf16* wfb   = (__bf16*)carve((size_t)16 * 64 * 2);
  float*  stats = (float*)carve(128 * 4);

  const int B = 256;

  // --- input / weight conversion ---
  {
    int n = N * 128;
    k_f32_to_bf16<<<(n + B - 1) / B, B, 0, stream>>>(x, hb, n);
    k_f32_to_bf16<<<(64 * 128 + B - 1) / B, B, 0, stream>>>(W1, w1b, 64 * 128);
    k_f32_to_bf16<<<(64 * 64 + B - 1) / B, B, 0, stream>>>(W2, w2b, 64 * 64);
    k_f32_to_bf16<<<(64 * 64 + B - 1) / B, B, 0, stream>>>(W3, w3b, 64 * 64);
    k_wf_pad_bf16<<<(16 * 64 + B - 1) / B, B, 0, stream>>>(Wf, wfb);
  }

  // --- degrees / normalization (graph fixed across layers) ---
  k_deg_init<<<(N + B - 1) / B, B, 0, stream>>>(dinv, N);
  k_deg_count<<<(E + B - 1) / B, B, 0, stream>>>(dst, dinv, E);
  k_deg_rsqrt<<<(N + B - 1) / B, B, 0, stream>>>(dinv, N);

  const int rowTiles   = (N + 15) / 16;
  const int gemmBlocks = (rowTiles + 7) / 8;     // 8 waves (row tiles) per block
  const int statBlocks = (N + 127) / 128;

  const float* bl[3]  = {b1, b2, b3};
  const float* gl[3]  = {g1, g2, g3};
  const float* bel[3] = {be1, be2, be3};

  for (int l = 0; l < 3; ++l) {
    // dense: h = H * W^T  (WMMA bf16 -> fp32; A loaded once per K-step per wave)
    if (l == 0)
      k_gemm_bf16<128, 4, 64, 64, false><<<gemmBlocks, 256, 0, stream>>>(hb, w1b, nullptr, h, N);
    else
      k_gemm_bf16<64, 4, 64, 64, false><<<gemmBlocks, 256, 0, stream>>>(hb, (l == 1) ? w2b : w3b, nullptr, h, N);
    // self-loop + bias init of aggregation buffer
    k_agg_init<<<(N * 16 + B - 1) / B, B, 0, stream>>>(h, dinv, bl[l], agg, N);
    // edge scatter-add (L2-resident h -> atomic f32 into agg)
    k_edge_agg<<<(int)(((long long)E * 16 + B - 1) / B), B, 0, stream>>>(src, dst, dinv, h, agg, E);
    // batchnorm stats + normalize + GELU, emit bf16 activations (stride 64)
    k_zero128<<<1, 128, 0, stream>>>(stats);
    k_bn_stats<<<statBlocks, 256, 0, stream>>>(agg, stats, N);
    k_bn_gelu<<<(N * 64 + B - 1) / B, B, 0, stream>>>(agg, stats, gl[l], bel[l], hb, N);
  }

  // final classifier: out = h3 * Wf^T + bf   (WMMA, 16-col padded, store 10)
  k_gemm_bf16<64, 1, 10, 10, true><<<gemmBlocks, 256, 0, stream>>>(hb, wfb, bf, out, N);
}
